// ParticleNet_31731218382958
// MI455X (gfx1250) — matrix operand order repaired and checked by reading, not compiled, not run on hardware
//
#include <hip/hip_runtime.h>
#include <hip/hip_bf16.h>
#include <math.h>

// ---------------------------------------------------------------------------
// ParticleNet inference for MI455X (gfx1250), wave32 + WMMA f16 16x16x32.
//
//   H (B*N, 451) f32; concat via descending channel offsets:
//   x@448, out0@384, out1@256, out2@0 -> each stage's h is a contiguous suffix.
//   Per conv: pack(f16,pad32) -> knn(top-16, LDS) -> pq (WMMA per-point GEMMs,
//   layer-1 factored as [xi|xj-xi]@W1^T = (P-Q)_i + Q_j) -> fused edge block
//   (4 points = 4 WMMA M-tiles per workgroup; B tiles loaded once, reused x4).
//   head: mean over N + fc1 + fc2 + sigmoid.
// ---------------------------------------------------------------------------

typedef _Float16 h8   __attribute__((ext_vector_type(8)));
typedef _Float16 v16h __attribute__((ext_vector_type(16)));
typedef float    v8f  __attribute__((ext_vector_type(8)));

union V16 { v16h v; h8 h[2]; };

// A-matrix (16x32 f16): lane l holds row (l&15); halves a[0..7] = K {0..7}
// (lanes<16) or {8..15} (lanes>=16); a[8..15] = same +16.
__device__ __forceinline__ v16h load_A(const _Float16* row, int lane) {
  V16 u;
  const int hi = (lane & 16) ? 8 : 0;
  u.h[0] = *(const h8*)(row + hi);
  u.h[1] = *(const h8*)(row + 16 + hi);
  return u.v;
}

// B-matrix (32x16 f16): lane l holds column (l&15); halves b[0..7] = K {0..7}
// (lanes<16) or {8..15} (lanes>=16); b[8..15] = same +16. `row` = weight row (B=W^T).
__device__ __forceinline__ v16h load_B(const _Float16* row, int lane) {
  V16 u;
  const int ko = (lane & 16) ? 8 : 0;
  u.h[0] = *(const h8*)(row + ko);
  u.h[1] = *(const h8*)(row + 16 + ko);
  return u.v;
}

__device__ __forceinline__ v8f wmma_f16(v16h a, v16h b, v8f c) {
  return __builtin_amdgcn_wmma_f32_16x16x32_f16(false, a, false, b,
                                                (short)0, c, false, false);
}

// ---------------------------------------------------------------------------
// Prep kernels
// ---------------------------------------------------------------------------

__global__ void k_scatter_x(const float* __restrict__ x, float* __restrict__ H) {
  int t = blockIdx.x * blockDim.x + threadIdx.x;
  if (t >= 16384 * 3) return;
  int m = t / 3, c = t - m * 3;
  H[m * 451 + 448 + c] = x[t];
}

// Fold BN into per-channel scale/bias: y = s*acc + t with
// s = gamma*rsqrt(var+eps), t = s*(b-mean)+beta  (linear bias folded in).
__global__ void k_fold(const float* __restrict__ b, const float* __restrict__ gamma,
                       const float* __restrict__ beta, const float* __restrict__ mean,
                       const float* __restrict__ var, float* __restrict__ slot, int n) {
  int c = blockIdx.x * blockDim.x + threadIdx.x;
  if (c >= n) return;
  float s = gamma[c] * rsqrtf(var[c] + 1e-5f);
  slot[c] = s;
  slot[256 + c] = s * (b[c] - mean[c]) + beta[c];
}

// Convert an f32 weight slice to f16 with zero padding to dstStride columns.
__global__ void k_cvt(const float* __restrict__ W, _Float16* __restrict__ dst,
                      int rows, int srcCols, int colOff, int copyCols, int dstStride) {
  int t = blockIdx.x * blockDim.x + threadIdx.x;
  if (t >= rows * dstStride) return;
  int r = t / dstStride, c = t - r * dstStride;
  dst[t] = (c < copyCols) ? (_Float16)W[r * srcCols + colOff + c] : (_Float16)0.0f;
}

// Pack h channel-suffix into f16, padded to inPad channels.
__global__ void k_pack(const float* __restrict__ H, _Float16* __restrict__ Hf,
                       int offIn, int Fin, int inPad) {
  int t = blockIdx.x * blockDim.x + threadIdx.x;
  if (t >= 16384 * inPad) return;
  int m = t / inPad, c = t - m * inPad;
  Hf[t] = (c < Fin) ? (_Float16)H[m * 451 + offIn + c] : (_Float16)0.0f;
}

// ---------------------------------------------------------------------------
// kNN: one block per batch, one thread per point, features staged in LDS.
// ---------------------------------------------------------------------------
__global__ __launch_bounds__(128) void k_knn(const _Float16* __restrict__ Hf,
                                             int inPad, int Fk,
                                             int* __restrict__ idxOut) {
  extern __shared__ _Float16 sf[];
  const int b = blockIdx.x, i = threadIdx.x;
  const _Float16* src = Hf + (size_t)b * 128 * inPad;
  for (int t = i; t < 128 * inPad; t += 128) sf[t] = src[t];
  __syncthreads();

  float bd[16]; int bi[16];
#pragma unroll
  for (int k = 0; k < 16; ++k) { bd[k] = 3.0e38f; bi[k] = 0; }

  for (int j = 0; j < 128; ++j) {
    float d;
    if (j == i) {
      d = 1.0e9f;                      // reference adds 1e9 on the diagonal
    } else {
      d = 0.0f;
      for (int c = 0; c < Fk; ++c) {
        float df = (float)sf[i * inPad + c] - (float)sf[j * inPad + c];
        d = fmaf(df, df, d);
      }
    }
    if (d < bd[15]) {                  // insert, keep ascending order
      int p = 15;
      while (p > 0 && bd[p - 1] > d) { bd[p] = bd[p - 1]; bi[p] = bi[p - 1]; --p; }
      bd[p] = d; bi[p] = j;
    }
  }
#pragma unroll
  for (int k = 0; k < 16; ++k) idxOut[(b * 128 + i) * 16 + k] = bi[k];
}

// ---------------------------------------------------------------------------
// Per-point GEMMs: P = x@W1a^T, Q = x@W1b^T. Writes PmQ = P-Q and Q.
// One block = one M-tile of 16 points; A fragments hoisted into registers and
// reused across all output N-tiles (they are nt-invariant).
// ---------------------------------------------------------------------------
template <int OUT, int INPAD>
__global__ __launch_bounds__(128) void k_pq(const _Float16* __restrict__ Hf,
                                            const _Float16* __restrict__ W1a,
                                            const _Float16* __restrict__ W1b,
                                            float* __restrict__ PmQ,
                                            float* __restrict__ Qb) {
  const int lane = threadIdx.x & 31;
  const int wave = threadIdx.x >> 5;
  const int mbase = blockIdx.x * 16;
  const _Float16* arow = Hf + (mbase + (lane & 15)) * INPAD;
  constexpr int NT = OUT / 16;
  constexpr int KS = INPAD / 32;

  v16h aT[KS];
#pragma unroll
  for (int ks = 0; ks < KS; ++ks) aT[ks] = load_A(arow + ks * 32, lane);

  for (int nt = wave; nt < NT; nt += 4) {
    const int c = nt * 16 + (lane & 15);
    const _Float16* brA = W1a + c * INPAD;
    const _Float16* brB = W1b + c * INPAD;
    v8f accP = {}; v8f accQ = {};
#pragma unroll
    for (int ks = 0; ks < KS; ++ks) {
      v16h ba = load_B(brA + ks * 32, lane);
      v16h bb = load_B(brB + ks * 32, lane);
      accP = wmma_f16(aT[ks], ba, accP);
      accQ = wmma_f16(aT[ks], bb, accQ);
    }
    const int mo = (lane & 16) ? 8 : 0;
#pragma unroll
    for (int r = 0; r < 8; ++r) {
      int m = mbase + mo + r;
      PmQ[m * OUT + c] = accP[r] - accQ[r];
      Qb [m * OUT + c] = accQ[r];
    }
  }
}

// ---------------------------------------------------------------------------
// Fused edge block: MT=4 points per block (4 WMMA M-tiles, 64 edge rows).
// Each B tile (weights) is loaded once per (nt,ks) and reused across the 4
// M-tiles -> 4x less L2 weight traffic than one-point-per-block.
//   L1 (VALU): X1[row][c] = relu(s1*(PmQ[pt]+Q[idx])+t1)     -> LDS f16
//   L2 (WMMA): X2 = relu(s2*(X1@W2^T)+t2)                    -> LDS f16
//   L3 (WMMA): y = relu(s3*(X2@W3^T)+t3); mean over 16 rows  -> H[offOut..]
// ---------------------------------------------------------------------------
template <int OUT>
__global__ __launch_bounds__(128) void k_edge(const float* __restrict__ PmQ,
                                              const float* __restrict__ Qb,
                                              const int* __restrict__ idx,
                                              const _Float16* __restrict__ W2,
                                              const _Float16* __restrict__ W3,
                                              const float* __restrict__ st1,
                                              const float* __restrict__ st2,
                                              const float* __restrict__ st3,
                                              float* __restrict__ H, int offOut) {
  constexpr int MT = 4;                       // points (M-tiles) per block
  constexpr int XS = OUT + 8;                 // LDS row stride (bank-conflict pad)
  __shared__ __align__(16) _Float16 Xa[64 * XS];
  __shared__ __align__(16) _Float16 Xb[64 * XS];
  __shared__ int nidx[64];

  const int tid = threadIdx.x;
  const int lane = tid & 31, wave = tid >> 5;
  const int p0 = blockIdx.x * MT;             // 4 consecutive points, same batch
  const int bb = p0 >> 7;

  if (tid < 64) nidx[tid] = idx[p0 * 16 + tid];
  __syncthreads();

  const float* s1 = st1; const float* t1 = st1 + 256;
  const float* s2 = st2; const float* t2 = st2 + 256;
  const float* s3 = st3; const float* t3 = st3 + 256;

  // Layer 1: assemble edge activations for all 64 rows.
  for (int v = tid; v < 64 * OUT; v += 128) {
    int row = v / OUT, c = v - row * OUT;     // row = mt*16 + k
    int pt = p0 + (row >> 4);
    float val = PmQ[pt * OUT + c] + Qb[((bb << 7) + nidx[row]) * OUT + c];
    float y = s1[c] * val + t1[c];
    Xa[row * XS + c] = (_Float16)(y > 0.0f ? y : 0.0f);
  }
  __syncthreads();

  constexpr int NT = OUT / 16;
  constexpr int KS = OUT / 32;
  const int mo = (lane & 16) ? 8 : 0;

  // Layer 2 (WMMA): B loaded once per (nt,ks), reused across 4 M-tiles.
  for (int nt = wave; nt < NT; nt += 4) {
    const int c = nt * 16 + (lane & 15);
    v8f acc[MT] = {};
#pragma unroll
    for (int ks = 0; ks < KS; ++ks) {
      v16h w = load_B(W2 + c * OUT + ks * 32, lane);
#pragma unroll
      for (int mt = 0; mt < MT; ++mt) {
        v16h a = load_A(&Xa[(mt * 16 + (lane & 15)) * XS + ks * 32], lane);
        acc[mt] = wmma_f16(a, w, acc[mt]);
      }
    }
#pragma unroll
    for (int mt = 0; mt < MT; ++mt) {
#pragma unroll
      for (int r = 0; r < 8; ++r) {
        float y = s2[c] * acc[mt][r] + t2[c];
        Xb[(mt * 16 + mo + r) * XS + c] = (_Float16)(y > 0.0f ? y : 0.0f);
      }
    }
  }
  __syncthreads();

  // Layer 3 (WMMA) + mean over each point's 16 neighbor rows.
  for (int nt = wave; nt < NT; nt += 4) {
    const int c = nt * 16 + (lane & 15);
    v8f acc[MT] = {};
#pragma unroll
    for (int ks = 0; ks < KS; ++ks) {
      v16h w = load_B(W3 + c * OUT + ks * 32, lane);
#pragma unroll
      for (int mt = 0; mt < MT; ++mt) {
        v16h a = load_A(&Xb[(mt * 16 + (lane & 15)) * XS + ks * 32], lane);
        acc[mt] = wmma_f16(a, w, acc[mt]);
      }
    }
#pragma unroll
    for (int mt = 0; mt < MT; ++mt) {
      float partial = 0.0f;
#pragma unroll
      for (int r = 0; r < 8; ++r) {
        float y = s3[c] * acc[mt][r] + t3[c];
        partial += (y > 0.0f ? y : 0.0f);
      }
      // lanes l and l^16 hold the same column, rows 0-7 / 8-15
      float tot = partial + __shfl_xor(partial, 16, 32);
      if (lane < 16) H[(p0 + mt) * 451 + offOut + nt * 16 + lane] = tot * (1.0f / 16.0f);
    }
  }
}

// ---------------------------------------------------------------------------
// Head: mean over N, fc1+relu, fc2, sigmoid. One block per batch element.
// ---------------------------------------------------------------------------
__global__ __launch_bounds__(256) void k_head(const float* __restrict__ H,
                                              const float* __restrict__ fc1W,
                                              const float* __restrict__ fc1b,
                                              const float* __restrict__ fc2W,
                                              const float* __restrict__ fc2b,
                                              float* __restrict__ out) {
  __shared__ float pooled[451];
  __shared__ float z[256];
  __shared__ float red[256];
  const int b = blockIdx.x, t = threadIdx.x;

  for (int c = t; c < 451; c += 256) {
    float s = 0.0f;
    for (int n = 0; n < 128; ++n) s += H[((b << 7) + n) * 451 + c];
    pooled[c] = s * (1.0f / 128.0f);
  }
  __syncthreads();

  {
    float d = fc1b[t];
    for (int c = 0; c < 451; ++c) d = fmaf(fc1W[t * 451 + c], pooled[c], d);
    z[t] = d > 0.0f ? d : 0.0f;
  }
  __syncthreads();

  red[t] = fc2W[t] * z[t];
  __syncthreads();
  for (int s = 128; s > 0; s >>= 1) {
    if (t < s) red[t] += red[t + s];
    __syncthreads();
  }
  if (t == 0) out[b] = 1.0f / (1.0f + expf(-(red[0] + fc2b[0])));
}

// ---------------------------------------------------------------------------
// Launch
// ---------------------------------------------------------------------------
static inline int wi(int conv, int layer, int field) {
  // inputs in setup_inputs() insertion order:
  // 0: x; then per conv/layer: W,b,gamma,beta,mean,var; then fc1_W,fc1_b,fc2_W,fc2_b
  return 1 + (conv * 3 + layer) * 6 + field;
}

extern "C" void kernel_launch(void* const* d_in, const int* in_sizes, int n_in,
                              void* d_out, int out_size, void* d_ws, size_t ws_size,
                              hipStream_t stream) {
  (void)in_sizes; (void)n_in; (void)out_size; (void)ws_size;

  static const int IN[3]     = {3, 67, 195};
  static const int OUT[3]    = {64, 128, 256};
  static const int INPAD[3]  = {32, 96, 224};
  static const int OFF_IN[3] = {448, 384, 256};
  static const int OFF_OUT[3]= {384, 256, 0};
  static const int FKNN[3]   = {2, 67, 195};

  char* ws = (char*)d_ws;
  float*     H    = (float*)    (ws);                         // 16384*451 f32
  _Float16*  Hf   = (_Float16*) (ws + 29556736);              // 16384*224 f16
  int*       idx  = (int*)      (ws + 36896768);              // 16384*16
  float*     PmQ  = (float*)    (ws + 37945344);              // 16384*256 f32
  float*     Qb   = (float*)    (ws + 54722560);              // 16384*256 f32
  _Float16*  W1a  = (_Float16*) (ws + 71499776);
  _Float16*  W1b  = (_Float16*) (ws + 71630848);
  _Float16*  W2f  = (_Float16*) (ws + 71761920);
  _Float16*  W3f  = (_Float16*) (ws + 71892992);
  float*     stF  = (float*)    (ws + 72024064);              // 9 layers * (s,t)

  const float* x = (const float*)d_in[0];
  float* out = (float*)d_out;

  // x -> H[..., 448:451]
  k_scatter_x<<<(16384 * 3 + 255) / 256, 256, 0, stream>>>(x, H);

  for (int i = 0; i < 3; ++i) {
    const int in = IN[i], outc = OUT[i], inPad = INPAD[i];

    // Fold BN (+linear bias) into scale/bias per layer.
    for (int j = 0; j < 3; ++j) {
      k_fold<<<1, 256, 0, stream>>>(
          (const float*)d_in[wi(i, j, 1)], (const float*)d_in[wi(i, j, 2)],
          (const float*)d_in[wi(i, j, 3)], (const float*)d_in[wi(i, j, 4)],
          (const float*)d_in[wi(i, j, 5)], stF + (i * 3 + j) * 512, outc);
    }

    // Weights -> f16 (W1 split into W1a|W1b, zero-padded to inPad).
    const float* W1 = (const float*)d_in[wi(i, 0, 0)];
    const float* W2 = (const float*)d_in[wi(i, 1, 0)];
    const float* W3 = (const float*)d_in[wi(i, 2, 0)];
    int n1 = outc * inPad, n2 = outc * outc;
    k_cvt<<<(n1 + 255) / 256, 256, 0, stream>>>(W1, W1a, outc, 2 * in, 0,  in, inPad);
    k_cvt<<<(n1 + 255) / 256, 256, 0, stream>>>(W1, W1b, outc, 2 * in, in, in, inPad);
    k_cvt<<<(n2 + 255) / 256, 256, 0, stream>>>(W2, W2f, outc, outc, 0, outc, outc);
    k_cvt<<<(n2 + 255) / 256, 256, 0, stream>>>(W3, W3f, outc, outc, 0, outc, outc);

    // Pack current h suffix to padded f16.
    k_pack<<<(16384 * inPad + 255) / 256, 256, 0, stream>>>(H, Hf, OFF_IN[i], in, inPad);

    // kNN on the packed features.
    k_knn<<<128, 128, 128 * inPad * 2, stream>>>(Hf, inPad, FKNN[i], idx);

    // Per-point GEMMs + fused edge MLP (4 points per block).
    const float* s1 = stF + (i * 3 + 0) * 512;
    const float* s2 = stF + (i * 3 + 1) * 512;
    const float* s3 = stF + (i * 3 + 2) * 512;
    if (i == 0) {
      k_pq<64, 32><<<1024, 128, 0, stream>>>(Hf, W1a, W1b, PmQ, Qb);
      k_edge<64><<<4096, 128, 0, stream>>>(PmQ, Qb, idx, W2f, W3f, s1, s2, s3, H, OFF_OUT[i]);
    } else if (i == 1) {
      k_pq<128, 96><<<1024, 128, 0, stream>>>(Hf, W1a, W1b, PmQ, Qb);
      k_edge<128><<<4096, 128, 0, stream>>>(PmQ, Qb, idx, W2f, W3f, s1, s2, s3, H, OFF_OUT[i]);
    } else {
      k_pq<256, 224><<<1024, 128, 0, stream>>>(Hf, W1a, W1b, PmQ, Qb);
      k_edge<256><<<4096, 128, 0, stream>>>(PmQ, Qb, idx, W2f, W3f, s1, s2, s3, H, OFF_OUT[i]);
    }
  }

  // Global mean pool + FC head.
  k_head<<<128, 256, 0, stream>>>(H, (const float*)d_in[55], (const float*)d_in[56],
                                  (const float*)d_in[57], (const float*)d_in[58], out);
}
